// SLURNNCRF_51221779972195
// MI455X (gfx1250) — matrix operand order repaired and checked by reading, not compiled, not run on hardware
//
#include <hip/hip_runtime.h>
#include <math.h>

#define BB 256
#define TT 128
#define EE 512
#define HP 256
#define GG 1024     // 4*Hp
#define KC 768      // E + Hp (concatenated GEMM K)
#define KK 74
#define KP 80       // padded K

typedef __attribute__((ext_vector_type(16))) __bf16 v16bf;
typedef __attribute__((ext_vector_type(8)))  float  v8f;

struct BF8x32 { unsigned int u[8]; };   // 8 dwords = 16 bf16 lane fragment

static __device__ __forceinline__ v16bf as_v16bf(BF8x32 r) {
  return __builtin_bit_cast(v16bf, r);
}
static __device__ __forceinline__ unsigned short f2bf(float f) {
  unsigned int u = __builtin_bit_cast(unsigned int, f);
  unsigned int r = u + 0x7fffu + ((u >> 16) & 1u);
  return (unsigned short)(r >> 16);
}
static __device__ __forceinline__ float sigmf(float x) {
  return 1.0f / (1.0f + expf(-x));
}

// ---------------------------------------------------------------------------
// One-time weight prep: bf16 [W_ih ; W_hh] concat per direction (n-major,
// k-contiguous so a WMMA B fragment is two 16B loads per lane), fused biases,
// padded w_out.
// ---------------------------------------------------------------------------
__global__ void prep_weights(const float* wihf, const float* whhf,
                             const float* bihf, const float* bhhf,
                             const float* wihb, const float* whhb,
                             const float* bihb, const float* bhhb,
                             const float* w_out, const float* b_out,
                             unsigned short* wcat, float* bias,
                             unsigned short* woutb, float* boutp) {
  int i = blockIdx.x * blockDim.x + threadIdx.x;
  if (i < 2 * GG * KC) {
    int dir = i / (GG * KC);
    int rem = i % (GG * KC);
    int n = rem / KC, k = rem % KC;
    const float* wih = dir ? wihb : wihf;
    const float* whh = dir ? whhb : whhf;
    float v = (k < EE) ? wih[n * EE + k] : whh[n * HP + (k - EE)];
    wcat[i] = f2bf(v);
  }
  if (i < 2 * GG) {
    int dir = i / GG, n = i % GG;
    bias[i] = dir ? (bihb[n] + bhhb[n]) : (bihf[n] + bhhf[n]);
  }
  if (i < KP * EE) {
    int n = i / EE, k = i % EE;
    woutb[i] = f2bf(n < KK ? w_out[n * EE + k] : 0.0f);
  }
  if (i < KP) boutp[i] = (i < KK) ? b_out[i] : 0.0f;
}

// ---------------------------------------------------------------------------
// Fused BiLSTM scan. Block = 16 batch rows, one direction. Per step:
// gather x_t (bf16 -> LDS), GEMM [x_t|h](16x768) @ Wcat(768x1024) via
// v_wmma_f32_16x16x32_bf16 (8 waves x 8 N-tiles, K-loop 24), gates -> LDS,
// pointwise LSTM with c/h in registers, masked carry, h back into LDS bf16.
// ---------------------------------------------------------------------------
#define SAP 784   // sA row stride in ushorts (768 + pad)

__global__ void __launch_bounds__(256) lstm_scan(
    const int* __restrict__ input_ids, const int* __restrict__ lengths,
    const float* __restrict__ emb, const unsigned short* __restrict__ wcat,
    const float* __restrict__ bias, float* __restrict__ hidden) {
  __shared__ __align__(16) unsigned short sA[16 * SAP];  // [row][x(512)|h(256)]
  __shared__ float sG[16 * GG];                          // gate pre-activations

  const int tid  = threadIdx.x;
  const int lane = tid & 31, wv = tid >> 5;
  const int half = lane >> 4, l16 = lane & 15;
  const int dir = blockIdx.y;
  const int b0  = blockIdx.x * 16;

  const unsigned short* W  = wcat + (size_t)dir * GG * KC;
  const float*          bs = bias + dir * GG;

  // pointwise ownership: row pr, 16 columns pc0 + 16*j
  const int pr  = tid >> 4;
  const int pc0 = tid & 15;
  float creg[16], hreg[16];
#pragma unroll
  for (int j = 0; j < 16; ++j) { creg[j] = 0.f; hreg[j] = 0.f; }
  const int mylen = lengths[b0 + pr];

  // zero h region of sA
#pragma unroll
  for (int j = 0; j < 16; ++j) sA[pr * SAP + EE + pc0 + 16 * j] = 0;

  for (int step = 0; step < TT; ++step) {
    const int t = dir ? (TT - 1 - step) : step;

    // stage x_t: thread covers 32 consecutive E-columns of its row
    {
      int tok = input_ids[(b0 + pr) * TT + t];
      const float* er = emb + (size_t)tok * EE + pc0 * 32;
#pragma unroll
      for (int j = 0; j < 8; ++j) {
        float4 v = ((const float4*)er)[j];
        unsigned short* d = &sA[pr * SAP + pc0 * 32 + 4 * j];
        d[0] = f2bf(v.x); d[1] = f2bf(v.y); d[2] = f2bf(v.z); d[3] = f2bf(v.w);
      }
    }
    __syncthreads();

    // GEMM: wave wv owns gate columns [wv*128, wv*128+128)
    v8f acc[8];
#pragma unroll
    for (int nt = 0; nt < 8; ++nt)
#pragma unroll
      for (int i = 0; i < 8; ++i) acc[nt][i] = 0.f;

    for (int kt = 0; kt < KC / 32; ++kt) {
      // A fragment: row = l16, contiguous K = half*16 + s (matches B)
      BF8x32 a;
#pragma unroll
      for (int j = 0; j < 8; ++j) {
        int kb = kt * 32 + half * 16 + j * 2;
        a.u[j] = *(const unsigned int*)&sA[l16 * SAP + kb];
      }
      v16bf av = as_v16bf(a);
#pragma unroll
      for (int nt = 0; nt < 8; ++nt) {
        int n = wv * 128 + nt * 16 + l16;
        const uint4* bp0 =
            (const uint4*)(W + (size_t)n * KC + kt * 32 + half * 16);
        BF8x32 bf;
        uint4 q0 = bp0[0];
        bf.u[0] = q0.x; bf.u[1] = q0.y; bf.u[2] = q0.z; bf.u[3] = q0.w;
        uint4 q1 = bp0[1];
        bf.u[4] = q1.x; bf.u[5] = q1.y; bf.u[6] = q1.z; bf.u[7] = q1.w;
        acc[nt] = __builtin_amdgcn_wmma_f32_16x16x32_bf16(
            false, av, false, as_v16bf(bf), (short)0, acc[nt], false, false);
      }
    }

    // scatter accumulators to gate buffer (C/D layout: elem i -> M=i+8*half)
#pragma unroll
    for (int nt = 0; nt < 8; ++nt) {
      int col = wv * 128 + nt * 16 + l16;
#pragma unroll
      for (int i = 0; i < 8; ++i) sG[(i + half * 8) * GG + col] = acc[nt][i];
    }
    __syncthreads();

    // pointwise LSTM cell
#pragma unroll
    for (int j = 0; j < 16; ++j) {
      int c = pc0 + 16 * j;
      float gi = sG[pr * GG + c]            + bs[c];
      float gf = sG[pr * GG + HP + c]       + bs[HP + c];
      float gg = sG[pr * GG + 2 * HP + c]   + bs[2 * HP + c];
      float go = sG[pr * GG + 3 * HP + c]   + bs[3 * HP + c];
      float cn = sigmf(gf) * creg[j] + sigmf(gi) * tanhf(gg);
      float hn = sigmf(go) * tanhf(cn);
      float m  = (t < mylen) ? 1.0f : 0.0f;
      hidden[((size_t)(b0 + pr) * TT + t) * 512 + dir * HP + c] = hn * m;
      if (m > 0.f) { creg[j] = cn; hreg[j] = hn; }
      sA[pr * SAP + EE + c] = f2bf(hreg[j]);
    }
    __syncthreads();
  }
}

// ---------------------------------------------------------------------------
// logits = hidden(32768x512) @ w_out^T (padded to 80) + b_out, WMMA bf16.
// ---------------------------------------------------------------------------
#define SHP 40
__global__ void __launch_bounds__(256) logits_gemm(
    const float* __restrict__ hidden, const unsigned short* __restrict__ woutb,
    const float* __restrict__ boutp, float* __restrict__ logits) {
  __shared__ __align__(16) unsigned short sH[128 * SHP];
  const int tid  = threadIdx.x;
  const int lane = tid & 31, wv = tid >> 5;
  const int half = lane >> 4, l16 = lane & 15;
  const size_t r0 = (size_t)blockIdx.x * 128;

  v8f acc[5];
#pragma unroll
  for (int nt = 0; nt < 5; ++nt)
#pragma unroll
    for (int i = 0; i < 8; ++i) acc[nt][i] = 0.f;

  for (int kt = 0; kt < EE / 32; ++kt) {
    __syncthreads();
    {  // stage 128 rows x 32 K-cols, fp32 -> bf16
      int rr = tid >> 1;
      int cc = (tid & 1) * 16;
      const float* src = hidden + (r0 + rr) * EE + kt * 32 + cc;
#pragma unroll
      for (int j = 0; j < 4; ++j) {
        float4 v = ((const float4*)src)[j];
        unsigned short* d = &sH[rr * SHP + cc + 4 * j];
        d[0] = f2bf(v.x); d[1] = f2bf(v.y); d[2] = f2bf(v.z); d[3] = f2bf(v.w);
      }
    }
    __syncthreads();
    BF8x32 a;
#pragma unroll
    for (int j = 0; j < 8; ++j) {
      int kb = half * 16 + j * 2;
      a.u[j] = *(const unsigned int*)&sH[(wv * 16 + l16) * SHP + kb];
    }
    v16bf av = as_v16bf(a);
#pragma unroll
    for (int nt = 0; nt < 5; ++nt) {
      int n = nt * 16 + l16;
      const uint4* bp0 = (const uint4*)(woutb + n * EE + kt * 32 + half * 16);
      BF8x32 bf;
      uint4 q0 = bp0[0];
      bf.u[0] = q0.x; bf.u[1] = q0.y; bf.u[2] = q0.z; bf.u[3] = q0.w;
      uint4 q1 = bp0[1];
      bf.u[4] = q1.x; bf.u[5] = q1.y; bf.u[6] = q1.z; bf.u[7] = q1.w;
      acc[nt] = __builtin_amdgcn_wmma_f32_16x16x32_bf16(
          false, av, false, as_v16bf(bf), (short)0, acc[nt], false, false);
    }
  }
#pragma unroll
  for (int nt = 0; nt < 5; ++nt) {
    int col = nt * 16 + l16;
#pragma unroll
    for (int i = 0; i < 8; ++i) {
      size_t row = r0 + wv * 16 + i + half * 8;
      logits[row * KP + col] = acc[nt][i] + boutp[col];
    }
  }
}

// ---------------------------------------------------------------------------
// CRF negative log-likelihood: one block per batch element.
// ---------------------------------------------------------------------------
__global__ void __launch_bounds__(128) crf_nll(
    const float* __restrict__ logits, const int* __restrict__ tag_ids,
    const int* __restrict__ lengths, const float* __restrict__ start,
    const float* __restrict__ endt, const float* __restrict__ trans,
    float* __restrict__ partial) {
  __shared__ float sT[KK * KK];
  __shared__ float aA[KK], aB[KK];
  __shared__ float red[128];
  __shared__ float sLogZ;
  const int b = blockIdx.x, tid = threadIdx.x;
  const int len = lengths[b];
  for (int i = tid; i < KK * KK; i += 128) sT[i] = trans[i];
  if (tid < KK) aA[tid] = start[tid] + logits[(size_t)b * TT * KP + tid];
  __syncthreads();
  float* cur = aA; float* nxt = aB;
  for (int t = 1; t < TT; ++t) {
    if (tid < KK) {
      float em = logits[((size_t)b * TT + t) * KP + tid];
      float m1 = -1e30f;
      for (int k = 0; k < KK; ++k)
        m1 = fmaxf(m1, cur[k] + sT[k * KK + tid]);
      float s = 0.f;
      for (int k = 0; k < KK; ++k)
        s += expf(cur[k] + sT[k * KK + tid] - m1);
      float nv = m1 + logf(s) + em;
      nxt[tid] = (t < len) ? nv : cur[tid];
    }
    __syncthreads();
    float* tmp = cur; cur = nxt; nxt = tmp;
  }
  if (tid == 0) {
    float m1 = -1e30f;
    for (int k = 0; k < KK; ++k) m1 = fmaxf(m1, cur[k] + endt[k]);
    float s = 0.f;
    for (int k = 0; k < KK; ++k) s += expf(cur[k] + endt[k] - m1);
    sLogZ = m1 + logf(s);
  }
  const int* tg = tag_ids + (size_t)b * TT;
  float acc = 0.f;
  if (tid == 0) {
    acc += start[tg[0]] + logits[(size_t)b * TT * KP + tg[0]];
    acc += endt[tg[len - 1]];
  }
  for (int t = 1 + tid; t < TT; t += 128)
    if (t < len)
      acc += sT[tg[t - 1] * KK + tg[t]] + logits[((size_t)b * TT + t) * KP + tg[t]];
  red[tid] = acc;
  __syncthreads();
  for (int s = 64; s > 0; s >>= 1) {
    if (tid < s) red[tid] += red[tid + s];
    __syncthreads();
  }
  if (tid == 0) partial[b] = red[0] - sLogZ;
}

// ---------------------------------------------------------------------------
// Viterbi decode: one block per batch element; backpointers in workspace.
// ---------------------------------------------------------------------------
__global__ void __launch_bounds__(128) viterbi(
    const float* __restrict__ logits, const int* __restrict__ lengths,
    const float* __restrict__ start, const float* __restrict__ endt,
    const float* __restrict__ trans, int* __restrict__ bp,
    float* __restrict__ pred) {
  __shared__ float sT[KK * KK];
  __shared__ float aA[KK], aB[KK];
  __shared__ int tags[TT];
  const int b = blockIdx.x, tid = threadIdx.x;
  const int len = lengths[b];
  int* mybp = bp + (size_t)b * (TT - 1) * KP;
  for (int i = tid; i < KK * KK; i += 128) sT[i] = trans[i];
  if (tid < KK) aA[tid] = start[tid] + logits[(size_t)b * TT * KP + tid];
  __syncthreads();
  float* cur = aA; float* nxt = aB;
  for (int t = 1; t < TT; ++t) {
    if (tid < KK) {
      float em = logits[((size_t)b * TT + t) * KP + tid];
      float best = -1e30f; int arg = 0;
      for (int k = 0; k < KK; ++k) {
        float v = cur[k] + sT[k * KK + tid];
        if (v > best) { best = v; arg = k; }
      }
      mybp[(t - 1) * KP + tid] = arg;
      nxt[tid] = (t < len) ? (best + em) : cur[tid];
    }
    __syncthreads();
    float* tmp = cur; cur = nxt; nxt = tmp;
  }
  if (tid == 0) {
    float best = -1e30f; int last = 0;
    for (int k = 0; k < KK; ++k) {
      float v = cur[k] + endt[k];
      if (v > best) { best = v; last = k; }
    }
    tags[TT - 1] = last;
    int carry = last;
    for (int s = TT - 2; s >= 0; --s) {
      if ((s + 1) < len) carry = mybp[s * KP + carry];
      tags[s] = carry;
    }
  }
  __syncthreads();
  for (int t = tid; t < TT; t += 128)
    pred[(size_t)b * TT + t] = (t < len) ? (float)tags[t] : 0.0f;
}

__global__ void reduce_loss(const float* __restrict__ partial,
                            float* __restrict__ out) {
  __shared__ float red[256];
  int tid = threadIdx.x;
  red[tid] = partial[tid];
  __syncthreads();
  for (int s = 128; s > 0; s >>= 1) {
    if (tid < s) red[tid] += red[tid + s];
    __syncthreads();
  }
  if (tid == 0) out[0] = -red[0] / (float)BB;
}

// ---------------------------------------------------------------------------
extern "C" void kernel_launch(void* const* d_in, const int* in_sizes, int n_in,
                              void* d_out, int out_size, void* d_ws,
                              size_t ws_size, hipStream_t stream) {
  const int*   input_ids = (const int*)d_in[0];
  const int*   lengths   = (const int*)d_in[1];
  const int*   tag_ids   = (const int*)d_in[2];
  const float* emb   = (const float*)d_in[3];
  const float* wihf  = (const float*)d_in[4];
  const float* whhf  = (const float*)d_in[5];
  const float* bihf  = (const float*)d_in[6];
  const float* bhhf  = (const float*)d_in[7];
  const float* wihb  = (const float*)d_in[8];
  const float* whhb  = (const float*)d_in[9];
  const float* bihb  = (const float*)d_in[10];
  const float* bhhb  = (const float*)d_in[11];
  const float* w_out = (const float*)d_in[12];
  const float* b_out = (const float*)d_in[13];
  const float* start = (const float*)d_in[14];
  const float* endt  = (const float*)d_in[15];
  const float* trans = (const float*)d_in[16];
  float* out = (float*)d_out;

  char* ws = (char*)d_ws;
  size_t off = 0;
  auto alloc = [&](size_t bytes) -> void* {
    off = (off + 255) & ~(size_t)255;
    void* p = ws + off;
    off += bytes;
    return p;
  };
  unsigned short* wcat  = (unsigned short*)alloc((size_t)2 * GG * KC * 2);
  float*          bias  = (float*)alloc((size_t)2 * GG * 4);
  unsigned short* woutb = (unsigned short*)alloc((size_t)KP * EE * 2);
  float*          boutp = (float*)alloc((size_t)KP * 4);
  float*          hidden = (float*)alloc((size_t)BB * TT * 512 * 4);
  float*          logitsb = (float*)alloc((size_t)BB * TT * KP * 4);
  int*            bp = (int*)alloc((size_t)BB * (TT - 1) * KP * 4);
  float*          partial = (float*)alloc((size_t)BB * 4);

  prep_weights<<<(2 * GG * KC + 255) / 256, 256, 0, stream>>>(
      wihf, whhf, bihf, bhhf, wihb, whhb, bihb, bhhb, w_out, b_out,
      wcat, bias, woutb, boutp);

  dim3 lg(BB / 16, 2);
  lstm_scan<<<lg, 256, 0, stream>>>(input_ids, lengths, emb, wcat, bias, hidden);

  logits_gemm<<<(BB * TT) / 128, 256, 0, stream>>>(hidden, woutb, boutp, logitsb);

  crf_nll<<<BB, 128, 0, stream>>>(logitsb, tag_ids, lengths, start, endt,
                                  trans, partial);
  viterbi<<<BB, 128, 0, stream>>>(logitsb, lengths, start, endt, trans, bp, out);
  reduce_loss<<<1, 256, 0, stream>>>(partial, out + (size_t)BB * TT);
}
